// CCSequenceModel_77610059038883
// MI455X (gfx1250) — matrix operand order repaired and checked by reading, not compiled, not run on hardware
//
#include <hip/hip_runtime.h>

// ---------------------------------------------------------------------------
// CDNA5 (gfx1250) fused GRU encoder-decoder sequence model.
//   B=4096, T_IN=256, N_IN=4, H=64, T_OUT=180
// One workgroup (192 threads = 6 wave32) owns 16 batch rows for the whole
// sequence. Recurrent weights live in VGPRs as v_wmma_f32_16x16x32_bf16
// B-fragments (12 frags = 96 VGPRs/wave); hidden state lives in LDS
// (fp32 master + bf16 WMMA copy). Gate buffers are column-major so each
// C-tile writes back as 2x ds_store_b128. The layer-0 recurrent GEMM of
// step t+1 is software-pipelined into step t's last stage -> only 3
// barriers per encoder step (4 in the autoregressive decoder).
// ---------------------------------------------------------------------------

#define B_TOTAL 4096
#define T_IN    256
#define N_IN    4
#define H       64
#define G       192   // 3*H  (r,z,n gates packed)
#define T_OUT   180
#define BT      16    // batch rows per workgroup (one WMMA M-tile)
#define NTHREADS 192  // 6 waves, 2 N-tiles (of 12) per wave

typedef __attribute__((ext_vector_type(16))) __bf16 bf16x16;
typedef __attribute__((ext_vector_type(8)))  __bf16 bf16x8;
typedef __attribute__((ext_vector_type(8)))  float  f32x8;
typedef __attribute__((ext_vector_type(4)))  float  f32x4;

__device__ __forceinline__ float sigm_f(float x) {
    return 1.0f / (1.0f + __expf(-x));
}
__device__ __forceinline__ float tanh_f(float x) {
    float e = __expf(2.0f * x);
    return 1.0f - 2.0f / (e + 1.0f);
}

union AFrag { bf16x16 v; bf16x8 h2[2]; };
union CFrag { f32x8 v; f32x4 q[2]; };

// Build B-fragments (32x16, KxN) for one wave from a global fp32 weight
// matrix W (G x H, row-major): B[k][n] = W[nbase+n][k].
// ISA layout: lanes 0-15 hold K=0..7,16..23 (one row per 16-bit half-slot),
// lanes 16-31 hold K=8..15,24..31; N = lane & 15.
__device__ __forceinline__ void build_bfrags(const float* __restrict__ Wg,
                                             bf16x16 (&Bf)[2][2],
                                             int nt0, int lane) {
    const int n  = lane & 15;
    const int kh = (lane >> 4) * 8;
#pragma unroll
    for (int t = 0; t < 2; ++t) {
        const float* rowp = Wg + (size_t)((nt0 + t) * 16 + n) * H;
#pragma unroll
        for (int kt = 0; kt < 2; ++kt) {
            union { bf16x16 v; __bf16 e[16]; } f;
#pragma unroll
            for (int j = 0; j < 16; ++j)
                f.e[j] = (__bf16)rowp[kt * 32 + kh + (j & 7) + ((j >> 3) << 4)];
            Bf[t][kt] = f.v;
        }
    }
}

// Per-phase bias registers: bias[t] = biasVec[(nt0+t)*16 + lane&15]
__device__ __forceinline__ void load_bias(const float* __restrict__ bv,
                                          float (&bias)[2], int nt0, int lane) {
#pragma unroll
    for (int t = 0; t < 2; ++t)
        bias[t] = bv[(nt0 + t) * 16 + (lane & 15)];
}

// One wave computes its 2 N-tiles of out = A(16x64) x B(64xG) + bias.
// A from LDS bf16 state (16x64 row-major). Output LDS buffer is
// COLUMN-major: out[col*BT + row], so a C-tile writes 2x ds_store_b128.
// A-frag layout (16-bit 16x32): lanes 0-15: M=lane, halves 0-7 -> K base 0,
// halves 8-15 -> K base 16; lanes 16-31: M=lane-16, K bases 8 / 24.
__device__ __forceinline__ void gemm2(const __bf16* __restrict__ Ab,
                                      const bf16x16 (&Bf)[2][2],
                                      const float (&bias)[2],
                                      int nt0,
                                      float* __restrict__ outLds,
                                      int lane) {
    const int row  = lane & 15;
    const int kh   = lane >> 4;
    const int ncol = lane & 15;
    const __bf16* base = Ab + row * H;

    AFrag a0, a1;
    a0.h2[0] = *(const bf16x8*)(base +  0 + 8 * kh);
    a0.h2[1] = *(const bf16x8*)(base + 16 + 8 * kh);
    a1.h2[0] = *(const bf16x8*)(base + 32 + 8 * kh);
    a1.h2[1] = *(const bf16x8*)(base + 48 + 8 * kh);

#pragma unroll
    for (int t = 0; t < 2; ++t) {
        const int nb = (nt0 + t) * 16;
        const float b = bias[t];                     // all 8 C regs share N
        CFrag c;
        c.v = (f32x8){ b, b, b, b, b, b, b, b };
        c.v = __builtin_amdgcn_wmma_f32_16x16x32_bf16(
                false, a0.v, false, Bf[t][0], (short)0, c.v, false, false);
        c.v = __builtin_amdgcn_wmma_f32_16x16x32_bf16(
                false, a1.v, false, Bf[t][1], (short)0, c.v, false, false);
        // C/D layout: lane L, vgpr i -> M = i + 8*(L>>4), N = L&15
        float* o = outLds + (nb + ncol) * BT + 8 * kh;
        *(f32x4*)(o)     = c.q[0];
        *(f32x4*)(o + 4) = c.q[1];
    }
}

// Layer-0 update, encoder: gi computed inline from x_t (N_IN=4 MACs/gate).
__device__ __forceinline__ void upd_l0_enc(const float* __restrict__ gh,
                                           const float* __restrict__ bi0,
                                           const float* __restrict__ wx,
                                           const float* __restrict__ xt,
                                           float* __restrict__ hf,
                                           __bf16* __restrict__ hb, int tid) {
    for (int idx = tid; idx < BT * H; idx += NTHREADS) {
        const int m = idx & 15;
        const int j = idx >> 4;
        const float x0 = xt[m*4+0], x1 = xt[m*4+1], x2 = xt[m*4+2], x3 = xt[m*4+3];
        const int gr = j, gz = H + j, gn = 2 * H + j;
        const float gir = bi0[gr] + x0*wx[gr*4+0] + x1*wx[gr*4+1] + x2*wx[gr*4+2] + x3*wx[gr*4+3];
        const float giz = bi0[gz] + x0*wx[gz*4+0] + x1*wx[gz*4+1] + x2*wx[gz*4+2] + x3*wx[gz*4+3];
        const float gin = bi0[gn] + x0*wx[gn*4+0] + x1*wx[gn*4+1] + x2*wx[gn*4+2] + x3*wx[gn*4+3];
        const float r = sigm_f(gir + gh[gr * BT + m]);
        const float z = sigm_f(giz + gh[gz * BT + m]);
        const float n = tanh_f(gin + r * gh[gn * BT + m]);
        float h = hf[m * H + j];
        h = (1.0f - z) * n + z * h;
        hf[m * H + j] = h;
        hb[m * H + j] = (__bf16)h;
    }
}

// Layer-0 update, decoder: input dim 1 -> gi = bi0 + prev*wx.
__device__ __forceinline__ void upd_l0_dec(const float* __restrict__ gh,
                                           const float* __restrict__ bi0,
                                           const float* __restrict__ wx,
                                           const float* __restrict__ prevS,
                                           float* __restrict__ hf,
                                           __bf16* __restrict__ hb, int tid) {
    for (int idx = tid; idx < BT * H; idx += NTHREADS) {
        const int m = idx & 15;
        const int j = idx >> 4;
        const float p = prevS[m];
        const int gr = j, gz = H + j, gn = 2 * H + j;
        const float r = sigm_f(bi0[gr] + p * wx[gr] + gh[gr * BT + m]);
        const float z = sigm_f(bi0[gz] + p * wx[gz] + gh[gz * BT + m]);
        const float n = tanh_f(bi0[gn] + p * wx[gn] + r * gh[gn * BT + m]);
        float h = hf[m * H + j];
        h = (1.0f - z) * n + z * h;
        hf[m * H + j] = h;
        hb[m * H + j] = (__bf16)h;
    }
}

// Layer-1 update: both gi (WMMA) and gh1 (WMMA) buffers, column-major.
__device__ __forceinline__ void upd_l1(const float* __restrict__ gi,
                                       const float* __restrict__ gh,
                                       float* __restrict__ hf,
                                       __bf16* __restrict__ hb, int tid) {
    for (int idx = tid; idx < BT * H; idx += NTHREADS) {
        const int m = idx & 15;
        const int j = idx >> 4;
        const int gr = j, gz = H + j, gn = 2 * H + j;
        const float r = sigm_f(gi[gr * BT + m] + gh[gr * BT + m]);
        const float z = sigm_f(gi[gz * BT + m] + gh[gz * BT + m]);
        const float n = tanh_f(gi[gn * BT + m] + r * gh[gn * BT + m]);
        float h = hf[m * H + j];
        h = (1.0f - z) * n + z * h;
        hf[m * H + j] = h;
        hb[m * H + j] = (__bf16)h;
    }
}

__global__ __launch_bounds__(NTHREADS)
void gru_seq_kernel(const float* __restrict__ x,
                    const float* __restrict__ eWih0, const float* __restrict__ eWhh0,
                    const float* __restrict__ ebih0, const float* __restrict__ ebhh0,
                    const float* __restrict__ eWih1, const float* __restrict__ eWhh1,
                    const float* __restrict__ ebih1, const float* __restrict__ ebhh1,
                    const float* __restrict__ dWih0, const float* __restrict__ dWhh0,
                    const float* __restrict__ dbih0, const float* __restrict__ dbhh0,
                    const float* __restrict__ dWih1, const float* __restrict__ dWhh1,
                    const float* __restrict__ dbih1, const float* __restrict__ dbhh1,
                    const float* __restrict__ Won,  const float* __restrict__ bon,
                    const float* __restrict__ Wcv,  const float* __restrict__ bcv,
                    float* __restrict__ out) {
    __shared__ __align__(16) float  gi_s[G * BT];      // 12 KB, col-major (gi1)
    __shared__ __align__(16) float  gh_s[G * BT];      // 12 KB, col-major (gh0)
    __shared__ __align__(16) float  gh1_s[G * BT];     // 12 KB, col-major (gh1)
    __shared__ __align__(16) float  h0f[BT * H];       //  4 KB (fp32 master)
    __shared__ __align__(16) float  h1f[BT * H];       //  4 KB
    __shared__ __align__(16) __bf16 h0b[BT * H];       //  2 KB (WMMA A source)
    __shared__ __align__(16) __bf16 h1b[BT * H];       //  2 KB
    __shared__ float bi0[G];                           // layer-0 input bias
    __shared__ float wx[G * N_IN];                     // layer-0 input weights
    __shared__ float wonS[H], wcvS[H];
    __shared__ float xt[BT * N_IN];
    __shared__ float prevS[BT];

    const int tid   = threadIdx.x;
    const int lane  = tid & 31;
    const int wv    = tid >> 5;
    const int nt0   = wv * 2;             // this wave's first N-tile (of 12)
    const int bbase = blockIdx.x * BT;

    // ---- init states / stage heads + encoder params ------------------------
    for (int i = tid; i < BT * H; i += NTHREADS) {
        h0f[i] = 0.0f; h1f[i] = 0.0f;
        h0b[i] = (__bf16)0.0f; h1b[i] = (__bf16)0.0f;
    }
    if (tid < H) { wonS[tid] = Won[tid]; wcvS[tid] = Wcv[tid]; }
    for (int g = tid; g < G; g += NTHREADS) bi0[g] = ebih0[g];
    for (int i = tid; i < G * N_IN; i += NTHREADS) wx[i] = eWih0[i];

    // Recurrent weights -> VGPR-resident WMMA B-fragments (12 x 8 VGPRs).
    bf16x16 Bh0[2][2], Bi1[2][2], Bh1[2][2];
    build_bfrags(eWhh0, Bh0, nt0, lane);
    build_bfrags(eWih1, Bi1, nt0, lane);
    build_bfrags(eWhh1, Bh1, nt0, lane);
    float bBh0[2], bBi1[2], bBh1[2];       // step-invariant GEMM bias registers
    load_bias(ebhh0, bBh0, nt0, lane);
    load_bias(ebih1, bBi1, nt0, lane);
    load_bias(ebhh1, bBh1, nt0, lane);
    __syncthreads();

    // Pipeline preamble: gh0 for step 0 (h0 = 0 -> bias only, still exact).
    gemm2(h0b, Bh0, bBh0, nt0, gh_s, lane);

    // ========================= encoder: 256 steps ===========================
    // Stage 1: upd_l0 (gi0 inline)    Stage 2: gi1 + gh1 GEMMs
    // Stage 3: upd_l1 || gh0(t+1) GEMM (software pipelined)
    for (int t = 0; t < T_IN; ++t) {
        if (tid < BT * N_IN) {
            const int m = tid >> 2, i = tid & 3;
            xt[tid] = x[(size_t)(bbase + m) * T_IN * N_IN + t * N_IN + i];
        }
        if (tid < BT && (t + 1) < T_IN)   // prefetch next step's x slice
            __builtin_prefetch(&x[(size_t)(bbase + tid) * T_IN * N_IN + (t + 1) * N_IN], 0, 3);
        __syncthreads();                                   // (1) gh_s, xt ready

        upd_l0_enc(gh_s, bi0, wx, xt, h0f, h0b, tid);      // h0'
        __syncthreads();                                   // (2) h0b ready

        gemm2(h0b, Bi1, bBi1, nt0, gi_s,  lane);           // gi1 = h0'@Wih1^T+bih1
        gemm2(h1b, Bh1, bBh1, nt0, gh1_s, lane);           // gh1 = h1 @Whh1^T+bhh1
        __syncthreads();                                   // (3) gi_s, gh1_s ready

        upd_l1(gi_s, gh1_s, h1f, h1b, tid);                // h1'
        if (t + 1 < T_IN)
            gemm2(h0b, Bh0, bBh0, nt0, gh_s, lane);        // gh0 for step t+1
    }
    __syncthreads();

    // ---- phase switch: decoder weights -------------------------------------
    for (int g = tid; g < G; g += NTHREADS) {
        bi0[g] = dbih0[g];
        wx[g]  = dWih0[g];                  // (3H,1) input weights
    }
    if (tid < BT) prevS[tid] = 0.0f;
    build_bfrags(dWhh0, Bh0, nt0, lane);
    build_bfrags(dWih1, Bi1, nt0, lane);
    build_bfrags(dWhh1, Bh1, nt0, lane);
    load_bias(dbhh0, bBh0, nt0, lane);
    load_bias(dbih1, bBi1, nt0, lane);
    load_bias(dbhh1, bBh1, nt0, lane);
    __syncthreads();

    const float bon_v = bon[0], bcv_v = bcv[0];

    // Pipeline preamble: decoder gh0 for step 0 from d1 = encoder-final h0.
    gemm2(h0b, Bh0, bBh0, nt0, gh_s, lane);

    // ==================== decoder: 180 autoregressive steps =================
    for (int t = 0; t < T_OUT; ++t) {
        __syncthreads();                                   // (1) gh_s, prevS ready

        upd_l0_dec(gh_s, bi0, wx, prevS, h0f, h0b, tid);   // d1'
        __syncthreads();                                   // (2) h0b ready

        gemm2(h0b, Bi1, bBi1, nt0, gi_s,  lane);
        gemm2(h1b, Bh1, bBh1, nt0, gh1_s, lane);
        __syncthreads();                                   // (3) gi_s, gh1_s ready

        upd_l1(gi_s, gh1_s, h1f, h1b, tid);                // d2'
        if (t + 1 < T_OUT)
            gemm2(h0b, Bh0, bBh0, nt0, gh_s, lane);        // gh0 for step t+1
        __syncthreads();                                   // (4) h1f ready

        // heads: logit = d2.Won + bon ; cv = d2.Wcv + bcv ; prev <- cv
        if (tid < 2 * BT) {
            const int  m    = tid & (BT - 1);
            const bool iscv = tid >= BT;
            const float* wp = iscv ? wcvS : wonS;
            float acc = iscv ? bcv_v : bon_v;
#pragma unroll
            for (int j = 0; j < H; ++j) acc += h1f[m * H + j] * wp[j];
            const size_t o = iscv
                ? ((size_t)B_TOTAL * T_OUT + (size_t)(bbase + m) * T_OUT + t)
                : ((size_t)(bbase + m) * T_OUT + t);
            out[o] = acc;
            if (iscv) prevS[m] = acc;
        }
    }
}

extern "C" void kernel_launch(void* const* d_in, const int* in_sizes, int n_in,
                              void* d_out, int out_size, void* d_ws, size_t ws_size,
                              hipStream_t stream) {
    (void)in_sizes; (void)n_in; (void)out_size; (void)d_ws; (void)ws_size;
    const float* x     = (const float*)d_in[0];
    const float* eWih0 = (const float*)d_in[1];
    const float* eWhh0 = (const float*)d_in[2];
    const float* ebih0 = (const float*)d_in[3];
    const float* ebhh0 = (const float*)d_in[4];
    const float* eWih1 = (const float*)d_in[5];
    const float* eWhh1 = (const float*)d_in[6];
    const float* ebih1 = (const float*)d_in[7];
    const float* ebhh1 = (const float*)d_in[8];
    const float* dWih0 = (const float*)d_in[9];
    const float* dWhh0 = (const float*)d_in[10];
    const float* dbih0 = (const float*)d_in[11];
    const float* dbhh0 = (const float*)d_in[12];
    const float* dWih1 = (const float*)d_in[13];
    const float* dWhh1 = (const float*)d_in[14];
    const float* dbih1 = (const float*)d_in[15];
    const float* dbhh1 = (const float*)d_in[16];
    const float* Won   = (const float*)d_in[17];
    const float* bon   = (const float*)d_in[18];
    const float* Wcv   = (const float*)d_in[19];
    const float* bcv   = (const float*)d_in[20];

    dim3 grid(B_TOTAL / BT);   // 256 independent batch tiles
    dim3 block(NTHREADS);      // 6 wave32
    gru_seq_kernel<<<grid, block, 0, stream>>>(
        x, eWih0, eWhh0, ebih0, ebhh0, eWih1, eWhh1, ebih1, ebhh1,
        dWih0, dWhh0, dbih0, dbhh0, dWih1, dWhh1, dbih1, dbhh1,
        Won, bon, Wcv, bcv, (float*)d_out);
}
